// ROISampler_86646670230229
// MI455X (gfx1250) — compile-verified, matrix operand order checked by reading, not run on hardware
//
#include <hip/hip_runtime.h>
#include <hip/hip_bf16.h>

#define B_   32
#define N_   8192
#define M_   256
#define NT_  8448      // N_ + M_
#define NUM_ 512
#define NFG_ 128
#define TOT_ (B_ * NT_)

typedef __attribute__((ext_vector_type(16))) _Float16 v16h;
typedef __attribute__((ext_vector_type(8)))  float    v8f;
typedef __attribute__((ext_vector_type(4)))  int      v4i_t;

// Async global->LDS path (gfx1250): guard on builtin availability so the file
// compiles on toolchains without the async builtins (falls back to plain copy).
#if defined(__has_builtin)
#if __has_builtin(__builtin_amdgcn_global_load_async_to_lds_b128) && \
    __has_builtin(__builtin_amdgcn_s_wait_asynccnt)
#define USE_ASYNC_LDS 1
#endif
#endif

// ---------------------------------------------------------------- threefry
__device__ __forceinline__ unsigned rotl32(unsigned x, int d) {
    return (x << d) | (x >> (32 - d));
}

__device__ __forceinline__ void threefry2x32(unsigned k0, unsigned k1,
                                             unsigned x0, unsigned x1,
                                             unsigned* o0, unsigned* o1) {
    unsigned ks[3] = { k0, k1, 0x1BD11BDAu ^ k0 ^ k1 };
    const int rA[4] = {13, 15, 26, 6};
    const int rB[4] = {17, 29, 16, 24};
    x0 += ks[0]; x1 += ks[1];
    #pragma unroll
    for (int g = 0; g < 5; ++g) {
        const int* R = (g & 1) ? rB : rA;
        #pragma unroll
        for (int r = 0; r < 4; ++r) { x0 += x1; x1 = rotl32(x1, R[r]); x1 ^= x0; }
        unsigned i = (unsigned)(g + 1);
        x0 += ks[i % 3];
        x1 += ks[(i + 1) % 3] + i;
    }
    *o0 = x0; *o1 = x1;
}

// ---------------------------------------------------------------- matching
__global__ void __launch_bounds__(256)
match_kernel(const float* __restrict__ boxes, const float* __restrict__ gtb,
             const int* __restrict__ gtc,
             float* __restrict__ mbox, int* __restrict__ mcls,
             int* __restrict__ mgidx, int* __restrict__ flags,
             int* __restrict__ posCount) {
    __shared__ float4 g[M_];
    __shared__ float  garea[M_];
    __shared__ int    gcls[M_];
    __shared__ int    ginv[M_];

    const int tid = threadIdx.x;
    const int bid = blockIdx.x;
    const int b   = bid / (NT_ / 256);
    const int seg = bid % (NT_ / 256);

    if (tid < M_) {
        float4 q = reinterpret_cast<const float4*>(gtb)[(size_t)b * M_ + tid];
        g[tid]     = q;
        garea[tid] = (q.z - q.x) * (q.w - q.y);
        gcls[tid]  = gtc[b * M_ + tid];
        float mx   = fmaxf(fmaxf(q.x, q.y), fmaxf(q.z, q.w));
        ginv[tid]  = (mx < 0.0f) ? 1 : 0;
    }
    __syncthreads();

    const int i = seg * 256 + tid;
    float4 bx;
    if (i < N_) bx = reinterpret_cast<const float4*>(boxes)[(size_t)b * N_ + i];
    else        bx = g[i - N_];
    const float areab = (bx.z - bx.x) * (bx.w - bx.y);

    float best = -2.0f;
    int   bix  = 0;
    #pragma unroll 4
    for (int j = 0; j < M_; ++j) {
        float4 q = g[j];
        float iy0 = fmaxf(bx.x, q.x), ix0 = fmaxf(bx.y, q.y);
        float iy1 = fminf(bx.z, q.z), ix1 = fminf(bx.w, q.w);
        float inter = fmaxf(iy1 - iy0, 0.0f) * fmaxf(ix1 - ix0, 0.0f);
        float uni   = areab + garea[j] - inter;
        float iou   = (uni > 0.0f) ? inter / uni : 0.0f;
        if (ginv[j]) iou = -1.0f;
        if (iou > best) { best = iou; bix = j; }   // first-max kept (argmax)
    }

    const bool pos = best >= 0.5f;                       // FG_IOU_THR
    const bool neg = (best >= 0.0f) && (best < 0.5f);    // [BG_LOW, BG_HIGH)
    const bool bg  = !pos;                               // ignored set is empty

    const size_t o = (size_t)b * NT_ + i;
    reinterpret_cast<float4*>(mbox)[o] = bg ? make_float4(0.f, 0.f, 0.f, 0.f) : g[bix];
    mcls[o]  = bg ? 0  : gcls[bix];
    mgidx[o] = bg ? -1 : bix;
    flags[o] = (pos ? 1 : 0) | (neg ? 2 : 0);
    if (pos) atomicAdd(&posCount[b], 1);
}

// ---------------------------------------------------------------- rng + masked keys
__global__ void __launch_bounds__(256)
rng_kernel(const int* __restrict__ flags, float* __restrict__ r,
           float* __restrict__ keyP, float* __restrict__ keyN) {
    const int f = blockIdx.x * blockDim.x + threadIdx.x;
    if (f >= TOT_) return;
    const unsigned half = TOT_ / 2;
    const bool lo = (unsigned)f < half;
    unsigned x0 = lo ? (unsigned)f : (unsigned)f - half;
    unsigned x1 = lo ? (unsigned)f + half : (unsigned)f;
    unsigned o0, o1;
    threefry2x32(0u, 42u, x0, x1, &o0, &o1);     // jax.random.key(42)
    unsigned bits = lo ? o0 : o1;
    float rv = __uint_as_float((bits >> 9) | 0x3F800000u) - 1.0f;
    r[f] = rv;
    const int fl = flags[f];
    keyP[f] = (fl & 1) ? rv : -1.0f;
    keyN[f] = (fl & 2) ? rv : -1.0f;
}

// ---------------------------------------------------------------- WMMA rank-by-counting
// rank_i = #{ j : key_j > key_i }.  Comparison bits (exact f32 compares) are packed
// into 16x32 f16 A-tiles; B = all-ones; v_wmma_f32_16x16x32_f16 accumulates counts
// exactly in f32 (sums <= 8448 << 2^24).
__global__ void __launch_bounds__(128)
count_greater_wmma(const float* __restrict__ keys, int* __restrict__ cnt) {
    __shared__ float col[NT_];

    const int b   = blockIdx.x / (NT_ / 64);
    const int t64 = blockIdx.x % (NT_ / 64);
    const int tid = threadIdx.x;
    const float* kbase = keys + (size_t)b * NT_;

#ifdef USE_ASYNC_LDS
    {
        for (int idx = tid; idx < NT_ / 4; idx += 128) {
            v4i_t* gp = (v4i_t*)(kbase + idx * 4);   // generic, const dropped
            v4i_t* lp = (v4i_t*)(col + idx * 4);     // generic (LDS object)
            __builtin_amdgcn_global_load_async_to_lds_b128(
                (__attribute__((address_space(1))) v4i_t*)gp,
                (__attribute__((address_space(3))) v4i_t*)lp, 0, 0);
        }
        __builtin_amdgcn_s_wait_asynccnt(0);
    }
#else
    {
        const float4* k4 = reinterpret_cast<const float4*>(kbase);
        float4*       c4 = reinterpret_cast<float4*>(col);
        for (int idx = tid; idx < NT_ / 4; idx += 128) {
            if (idx + 128 < NT_ / 4) __builtin_prefetch(k4 + idx + 128, 0, 1);
            c4[idx] = k4[idx];
        }
    }
#endif
    __syncthreads();

    const int lane = tid & 31;
    const int wave = tid >> 5;
    const int Mr   = lane & 15;
    const int hi   = lane >> 4;
    const int rowbase = t64 * 64 + wave * 16;
    const float rowv  = col[rowbase + Mr];

    v8f acc = {0.f, 0.f, 0.f, 0.f, 0.f, 0.f, 0.f, 0.f};
    v16h ones;
    #pragma unroll
    for (int e = 0; e < 16; ++e) ones[e] = (_Float16)1.0f;

    const float2* c2 = reinterpret_cast<const float2*>(col);
    // Two K=32 tiles per iteration: all LDS reads issued up-front (one ds clause,
    // single wait), then VALU packing, then two back-to-back WMMAs.
    for (int jc = 0; jc < NT_; jc += 64) {
        float2 p0[8], p1[8];
        #pragma unroll
        for (int v = 0; v < 8; ++v) {
            // A-fragment 16x32 f16 layout: lane L holds M=L%16; K pairs per VGPR,
            // lanes 16-31 offset K by +8 within each 16-wide K half.
            const int kb = ((v < 4) ? 2 * v : 16 + 2 * (v - 4)) + hi * 8;
            p0[v] = c2[(jc + kb) >> 1];
            p1[v] = c2[(jc + 32 + kb) >> 1];
        }
        v16h a0, a1;
        #pragma unroll
        for (int v = 0; v < 8; ++v) {
            a0[2 * v]     = (p0[v].x > rowv) ? (_Float16)1.0f : (_Float16)0.0f;
            a0[2 * v + 1] = (p0[v].y > rowv) ? (_Float16)1.0f : (_Float16)0.0f;
            a1[2 * v]     = (p1[v].x > rowv) ? (_Float16)1.0f : (_Float16)0.0f;
            a1[2 * v + 1] = (p1[v].y > rowv) ? (_Float16)1.0f : (_Float16)0.0f;
        }
        acc = __builtin_amdgcn_wmma_f32_16x16x32_f16(
                  false, a0, false, ones, (short)0, acc, false, false);
        acc = __builtin_amdgcn_wmma_f32_16x16x32_f16(
                  false, a1, false, ones, (short)0, acc, false, false);
    }

    // D layout: VGPR g holds row M=g (lanes 0-15) / M=g+8 (lanes 16-31), count
    // replicated across all 16 N columns -> one writer per 16-lane half.
    if ((lane & 15) == 0) {
        const int base = rowbase + hi * 8;
        #pragma unroll
        for (int g2 = 0; g2 < 8; ++g2)
            cnt[(size_t)b * NT_ + base + g2] = (int)(acc[g2] + 0.5f);
    }
}

// ---------------------------------------------------------------- selection / score
__global__ void __launch_bounds__(256)
select_kernel(const int* __restrict__ flags, const int* __restrict__ cntP,
              const int* __restrict__ cntN, const int* __restrict__ posCount,
              const float* __restrict__ r, float* __restrict__ score) {
    const int f = blockIdx.x * blockDim.x + threadIdx.x;
    if (f >= TOT_) return;
    const int b  = f / NT_;
    const int fl = flags[f];
    const int np = min(posCount[b], NFG_);            // n_pos = sum(sel_pos)
    const bool selP = (fl & 1) && (cntP[f] < NFG_);
    const bool selN = (fl & 2) && (cntN[f] < (NUM_ - np));
    score[f] = ((selP || selN) ? 2.0f : 0.0f) + r[f];
}

// ---------------------------------------------------------------- gather by final rank
__global__ void __launch_bounds__(256)
gather_kernel(const int* __restrict__ cntS, const float* __restrict__ boxes,
              const float* __restrict__ gtb, const float* __restrict__ mbox,
              const int* __restrict__ mcls, const int* __restrict__ mgidx,
              float* __restrict__ out) {
    const int f = blockIdx.x * blockDim.x + threadIdx.x;
    if (f >= TOT_) return;
    const int rank = cntS[f];
    if (rank >= NUM_) return;                // ranks are a permutation: 512 hits
    const int b = f / NT_;
    const int i = f % NT_;

    float4 bx;
    if (i < N_) bx = reinterpret_cast<const float4*>(boxes)[(size_t)b * N_ + i];
    else        bx = reinterpret_cast<const float4*>(gtb)[(size_t)b * M_ + (i - N_)];

    float* rois = out;                         // (B,512,4)
    float* gbo  = out + (size_t)B_ * NUM_ * 4; // (B,512,4)
    float* cls  = out + (size_t)B_ * NUM_ * 8; // (B,512)
    float* gix  = cls + (size_t)B_ * NUM_;     // (B,512)

    const size_t p = (size_t)b * NUM_ + rank;
    reinterpret_cast<float4*>(rois)[p] = bx;
    reinterpret_cast<float4*>(gbo)[p]  = reinterpret_cast<const float4*>(mbox)[f];
    cls[p] = (float)mcls[f];
    gix[p] = (float)mgidx[f];
}

// ---------------------------------------------------------------- launcher
extern "C" void kernel_launch(void* const* d_in, const int* in_sizes, int n_in,
                              void* d_out, int out_size, void* d_ws, size_t ws_size,
                              hipStream_t stream) {
    const float* boxes = (const float*)d_in[0];   // (32,8192,4) f32
    const float* gtb   = (const float*)d_in[1];   // (32,256,4)  f32
    const int*   gtc   = (const int*)  d_in[2];   // (32,256)    i32
    float* out = (float*)d_out;

    // workspace carving (all 16B-aligned slabs of TOT_ elements)
    char* w = (char*)d_ws;
    float* mbox  = (float*)w;                 w += (size_t)TOT_ * 4 * sizeof(float);
    int*   mcls  = (int*)w;                   w += (size_t)TOT_ * sizeof(int);
    int*   mgidx = (int*)w;                   w += (size_t)TOT_ * sizeof(int);
    int*   flags = (int*)w;                   w += (size_t)TOT_ * sizeof(int);
    float* r     = (float*)w;                 w += (size_t)TOT_ * sizeof(float);
    float* keyP  = (float*)w;                 w += (size_t)TOT_ * sizeof(float);
    float* keyN  = (float*)w;                 w += (size_t)TOT_ * sizeof(float);
    int*   cntP  = (int*)w;                   w += (size_t)TOT_ * sizeof(int);
    int*   cntN  = (int*)w;                   w += (size_t)TOT_ * sizeof(int);
    float* score = (float*)w;                 w += (size_t)TOT_ * sizeof(float);
    int*   cntS  = (int*)w;                   w += (size_t)TOT_ * sizeof(int);
    int*   posCount = (int*)w;                w += 64 * sizeof(int);
    (void)ws_size; (void)in_sizes; (void)n_in; (void)out_size;

    (void)hipMemsetAsync(posCount, 0, B_ * sizeof(int), stream);

    match_kernel<<<B_ * (NT_ / 256), 256, 0, stream>>>(
        boxes, gtb, gtc, mbox, mcls, mgidx, flags, posCount);

    const int g1 = (TOT_ + 255) / 256;
    rng_kernel<<<g1, 256, 0, stream>>>(flags, r, keyP, keyN);

    const int gc = B_ * (NT_ / 64);
    count_greater_wmma<<<gc, 128, 0, stream>>>(keyP, cntP);
    count_greater_wmma<<<gc, 128, 0, stream>>>(keyN, cntN);

    select_kernel<<<g1, 256, 0, stream>>>(flags, cntP, cntN, posCount, r, score);

    count_greater_wmma<<<gc, 128, 0, stream>>>(score, cntS);

    gather_kernel<<<g1, 256, 0, stream>>>(cntS, boxes, gtb, mbox, mcls, mgidx, out);
}
